// NattenRegistersBlock_50388556317169
// MI455X (gfx1250) — compile-verified, matrix operand order checked by reading, not run on hardware
//
#include <hip/hip_runtime.h>

// ---------------------------------------------------------------------------
// NATTEN (7x7 neighborhood attention + 16 register tokens) for MI455X gfx1250
// All matmul stages on v_wmma_f32_16x16x32_bf16 (wave32 WMMA).
//  - GEMMs: 64x64 tile / wave (16 accumulators) for 4x operand reuse
//  - Attention: fixed 7x32 padded key window -> shift-only addressing,
//    V stored transposed (vT[h][e][pix]) so P*V B-fragments are contiguous
// ---------------------------------------------------------------------------

#define HH   64
#define WW   64
#define DM   512
#define NH   8
#define EE   64
#define KSZ  7
#define RR   16
#define PP   (HH * WW)       // 4096 pixels
#define PPP  (PP + 64)       // padded pixel stride (edge-window overrun)
#define NLOC 224             // 7 rows x 32 padded cols
#define NCOL 240             // + 16 register slots
#define SPC  256             // bf16 prob row stride (padded to 8 chunks of 32)

typedef __attribute__((ext_vector_type(16))) __bf16 v16bf;
typedef __attribute__((ext_vector_type(8)))  float  v8f;

union Frag {
  v16bf          v;
  unsigned int   u[8];
  unsigned short s[16];
};

// float -> bf16 round-to-nearest-even
__device__ __forceinline__ unsigned short f2bf(float f) {
  unsigned int u = __float_as_uint(f);
  u += 0x7FFFu + ((u >> 16) & 1u);
  return (unsigned short)(u >> 16);
}

// Per-lane K offset of dword v (0..7) in a 16-bit WMMA A/B fragment.
// Lanes 0-15 hold K={0..7,16..23} pairs, lanes 16-31 K={8..15,24..31} pairs.
__device__ __forceinline__ int kbase(int lane, int v) {
  return ((v >> 2) << 4) + ((lane & 16) ? 8 : 0) + ((v & 3) << 1);
}

// Load fragment whose per-lane row is contiguous in K (stride-1 bf16).
__device__ __forceinline__ void load_rowk(Frag& f, const unsigned short* rowptr,
                                          int lane) {
#pragma unroll
  for (int v = 0; v < 8; ++v)
    f.u[v] = *reinterpret_cast<const unsigned int*>(rowptr + kbase(lane, v));
}

__device__ __forceinline__ v8f wmma_bf16(const Frag& a, const Frag& b, v8f c) {
  return __builtin_amdgcn_wmma_f32_16x16x32_bf16(
      false, a.v, false, b.v, (short)0, c, false, false);
}

// ---------------------------------------------------------------------------
// Kernel 0: f32 -> bf16
// ---------------------------------------------------------------------------
__global__ void cvt_f32_bf16(const float* __restrict__ src,
                             unsigned short* __restrict__ dst, int n) {
  int i = blockIdx.x * blockDim.x + threadIdx.x;
  if (i < n) dst[i] = f2bf(src[i]);
}

// ---------------------------------------------------------------------------
// Kernel 1: QKV projection, 64x64 tile per wave.
// q -> qbuf[h][pix][e] (scaled), k -> kbuf[h][pix(padded)][e],
// v -> vT[h][e][pix(padded)]  (transposed for the attention AV stage)
// ---------------------------------------------------------------------------
__global__ __launch_bounds__(32)
void qkv_gemm(const unsigned short* __restrict__ xb,
              const unsigned short* __restrict__ wb,
              unsigned short* __restrict__ qbuf,
              unsigned short* __restrict__ kbuf,
              unsigned short* __restrict__ vT) {
  const int lane = threadIdx.x;
  const int m0   = blockIdx.x * 64;
  const int g    = blockIdx.y;          // 0..23
  const int t    = g >> 3;              // 0=q 1=k 2=v
  const int h    = g & 7;

  v8f acc[4][4];
#pragma unroll
  for (int ms = 0; ms < 4; ++ms)
#pragma unroll
    for (int s = 0; s < 4; ++s) acc[ms][s] = {};

  for (int kt = 0; kt < DM / 32; ++kt) {
    const int k0 = kt * 32;
    Frag a[4];
#pragma unroll
    for (int ms = 0; ms < 4; ++ms)
      load_rowk(a[ms], xb + (m0 + ms * 16 + (lane & 15)) * DM + k0, lane);
#pragma unroll
    for (int s = 0; s < 4; ++s) {
      Frag b;
      load_rowk(b, wb + (g * 64 + s * 16 + (lane & 15)) * DM + k0, lane);
#pragma unroll
      for (int ms = 0; ms < 4; ++ms)
        acc[ms][s] = wmma_bf16(a[ms], b, acc[ms][s]);
    }
  }

  const float scale = (t == 0) ? 0.125f : 1.0f;   // 1/sqrt(64) on Q
#pragma unroll
  for (int ms = 0; ms < 4; ++ms) {
#pragma unroll
    for (int s = 0; s < 4; ++s) {
      const int e = s * 16 + (lane & 15);
#pragma unroll
      for (int r = 0; r < 8; ++r) {
        const int m = m0 + ms * 16 + r + ((lane & 16) ? 8 : 0);
        const unsigned short val = f2bf(acc[ms][s][r] * scale);
        if (t == 0)       qbuf[((size_t)h * PP + m) * EE + e] = val;
        else if (t == 1)  kbuf[((size_t)h * PPP + m) * EE + e] = val;
        else              vT[((size_t)(h * EE + e)) * PPP + m] = val;
      }
    }
  }
}

// ---------------------------------------------------------------------------
// Kernel 2: register K/V projection. regk[h][r][e], regvT[h][e][r]
// ---------------------------------------------------------------------------
__global__ __launch_bounds__(32)
void regkv_gemm(const unsigned short* __restrict__ regb,
                const unsigned short* __restrict__ wb,   // full 1536x512
                unsigned short* __restrict__ regk,
                unsigned short* __restrict__ regvT) {
  const int lane = threadIdx.x;
  const int g    = blockIdx.x;          // 0..15
  const int t    = g >> 3;              // 0=k 1=v
  const int h    = g & 7;

  v8f acc[4];
#pragma unroll
  for (int s = 0; s < 4; ++s) acc[s] = {};

  const unsigned short* arow = regb + (lane & 15) * DM;
  for (int kt = 0; kt < DM / 32; ++kt) {
    const int k0 = kt * 32;
    Frag a; load_rowk(a, arow + k0, lane);
#pragma unroll
    for (int s = 0; s < 4; ++s) {
      Frag b;
      load_rowk(b, wb + (DM + g * 64 + s * 16 + (lane & 15)) * DM + k0, lane);
      acc[s] = wmma_bf16(a, b, acc[s]);
    }
  }

#pragma unroll
  for (int s = 0; s < 4; ++s) {
    const int e = s * 16 + (lane & 15);
#pragma unroll
    for (int r = 0; r < 8; ++r) {
      const int m = r + ((lane & 16) ? 8 : 0);
      if (t == 0) regk[(h * RR + m) * EE + e]  = f2bf(acc[s][r]);
      else        regvT[(h * EE + e) * RR + m] = f2bf(acc[s][r]);
    }
  }
}

// ---------------------------------------------------------------------------
// Kernel 3: neighborhood attention. One wave per (head, row, 16-query tile).
// Padded key window: 7 rows x 32 cols starting at even base ub -> 224 local
// slots + 16 registers. Logits via WMMA -> masked LDS, softmax, P*V via WMMA
// with contiguous B-fragments from vT.
// ---------------------------------------------------------------------------
__global__ __launch_bounds__(32)
void natten_attn(const unsigned short* __restrict__ qbuf,
                 const unsigned short* __restrict__ kbuf,
                 const unsigned short* __restrict__ vT,
                 const unsigned short* __restrict__ regk,
                 const unsigned short* __restrict__ regvT,
                 unsigned short* __restrict__ attn) {
  const int lane = threadIdx.x;
  const int x0   = blockIdx.x * 16;
  const int y    = blockIdx.y;
  const int h    = blockIdx.z;

  const int s_lo = min(max(x0 - 3, 0), WW - KSZ);
  const int ub   = s_lo & ~1;                      // even base, covers union
  const int ys   = min(max(y - 3, 0), HH - KSZ);

  __shared__ float          slog[16 * NCOL];
  __shared__ unsigned short sprob[16 * SPC];

  // Q fragments: 16 queries x 64 dims (pre-scaled)
  const unsigned short* qrow =
      qbuf + ((size_t)h * PP + (y * WW + x0 + (lane & 15))) * EE;
  Frag qa0, qa1;
  load_rowk(qa0, qrow, lane);
  load_rowk(qa1, qrow + 32, lane);

  // ---- logits: 14 local tiles + 1 register tile -------------------------
  for (int kt = 0; kt < 15; ++kt) {
    v8f acc = {};
    int kxx = 0;
    const unsigned short* krow;
    if (kt < 14) {
      const int idx = kt * 16 + (lane & 15);       // 0..223
      const int kr  = idx >> 5;                    // key row 0..6
      kxx = ub + (idx & 31);                       // key col (may pass masks)
      krow = kbuf + ((size_t)h * PPP + (ys + kr) * WW + kxx) * EE;
    } else {
      krow = regk + (h * RR + (lane & 15)) * EE;
    }
    Frag b;
    load_rowk(b, krow, lane);
    acc = wmma_bf16(qa0, b, acc);
    load_rowk(b, krow + 32, lane);
    acc = wmma_bf16(qa1, b, acc);

#pragma unroll
    for (int r = 0; r < 8; ++r) {
      const int q = r + ((lane & 16) ? 8 : 0);
      if (kt == 14) {
        slog[q * NCOL + NLOC + (lane & 15)] = acc[r];
      } else {
        const int  x  = x0 + q;
        const int  xs = min(max(x - 3, 0), WW - KSZ);
        const bool valid = (unsigned)(kxx - xs) < (unsigned)KSZ;
        slog[q * NCOL + kt * 16 + (lane & 15)] = valid ? acc[r] : -1e30f;
      }
    }
  }
  __syncthreads();

  // ---- softmax: one query per lane (lanes 0..15) ------------------------
  if (lane < 16) {
    float m = -1e30f;
    for (int j = 0; j < NCOL; ++j) m = fmaxf(m, slog[lane * NCOL + j]);
    float ssum = 0.f;
    for (int j = 0; j < NCOL; ++j) {
      const float e = __expf(slog[lane * NCOL + j] - m);
      slog[lane * NCOL + j] = e;
      ssum += e;
    }
    const float inv = 1.0f / ssum;
    for (int j = 0; j < NCOL; ++j)
      sprob[lane * SPC + j] = f2bf(slog[lane * NCOL + j] * inv);
    for (int j = NCOL; j < SPC; ++j) sprob[lane * SPC + j] = 0;
  }
  __syncthreads();

  // ---- out = P(16x256) x V(256x64): chunk kc = key row; kc==7 = registers
  v8f out[4];
#pragma unroll
  for (int s = 0; s < 4; ++s) out[s] = {};

  for (int kc = 0; kc < 8; ++kc) {
    Frag a;
    load_rowk(a, sprob + (lane & 15) * SPC + kc * 32, lane);
#pragma unroll
    for (int s = 0; s < 4; ++s) {
      const int eo = s * 16 + (lane & 15);
      Frag b;
      if (kc < 7) {
        const unsigned short* vrow =
            vT + ((size_t)(h * EE + eo)) * PPP + (ys + kc) * WW + ub;
        load_rowk(b, vrow, lane);                   // contiguous in key col
      } else {
        const unsigned short* rrow = regvT + (h * EE + eo) * RR;
#pragma unroll
        for (int v = 0; v < 4; ++v)                 // K0 in [0,16): registers
          b.u[v] = *reinterpret_cast<const unsigned int*>(rrow + kbase(lane, v));
#pragma unroll
        for (int v = 4; v < 8; ++v) b.u[v] = 0;     // K0 in [16,32): zero pad
      }
      out[s] = wmma_bf16(a, b, out[s]);
    }
  }

#pragma unroll
  for (int s = 0; s < 4; ++s) {
    const int e = s * 16 + (lane & 15);
#pragma unroll
    for (int r = 0; r < 8; ++r) {
      const int q   = r + ((lane & 16) ? 8 : 0);
      const int pix = y * WW + x0 + q;
      attn[(size_t)pix * DM + h * EE + e] = f2bf(out[s][r]);
    }
  }
}

// ---------------------------------------------------------------------------
// Kernel 4: output projection, 64x64 tile per wave, f32 out.
// ---------------------------------------------------------------------------
__global__ __launch_bounds__(32)
void out_gemm(const unsigned short* __restrict__ attn,
              const unsigned short* __restrict__ wob,
              float* __restrict__ out) {
  const int lane = threadIdx.x;
  const int m0   = blockIdx.x * 64;
  const int n0   = blockIdx.y * 64;

  v8f acc[4][4];
#pragma unroll
  for (int ms = 0; ms < 4; ++ms)
#pragma unroll
    for (int s = 0; s < 4; ++s) acc[ms][s] = {};

  for (int kt = 0; kt < DM / 32; ++kt) {
    const int k0 = kt * 32;
    Frag a[4];
#pragma unroll
    for (int ms = 0; ms < 4; ++ms)
      load_rowk(a[ms], attn + (size_t)(m0 + ms * 16 + (lane & 15)) * DM + k0,
                lane);
#pragma unroll
    for (int s = 0; s < 4; ++s) {
      Frag b;
      load_rowk(b, wob + (n0 + s * 16 + (lane & 15)) * DM + k0, lane);
#pragma unroll
      for (int ms = 0; ms < 4; ++ms)
        acc[ms][s] = wmma_bf16(a[ms], b, acc[ms][s]);
    }
  }

#pragma unroll
  for (int ms = 0; ms < 4; ++ms) {
#pragma unroll
    for (int s = 0; s < 4; ++s) {
      const int n = n0 + s * 16 + (lane & 15);
#pragma unroll
      for (int r = 0; r < 8; ++r) {
        const int m = m0 + ms * 16 + r + ((lane & 16) ? 8 : 0);
        out[(size_t)m * DM + n] = acc[ms][s][r];
      }
    }
  }
}

// ---------------------------------------------------------------------------
extern "C" void kernel_launch(void* const* d_in, const int* in_sizes, int n_in,
                              void* d_out, int out_size, void* d_ws,
                              size_t ws_size, hipStream_t stream) {
  (void)n_in; (void)out_size; (void)ws_size;
  const float* x    = (const float*)d_in[0];   // (1,64,64,512)
  const float* regs = (const float*)d_in[1];   // (16,512)
  const float* wqkv = (const float*)d_in[2];   // (1536,512)
  const float* wout = (const float*)d_in[3];   // (512,512)
  float* out = (float*)d_out;                  // (4096,512) f32

  char* ws = (char*)d_ws;
  size_t off = 0;
  auto carve = [&](size_t elems) {
    void* p = ws + off;
    off = (off + elems * sizeof(unsigned short) + 255) & ~(size_t)255;
    return (unsigned short*)p;
  };
  unsigned short* xb    = carve((size_t)PP * DM);        // 4096x512
  unsigned short* wqkvb = carve((size_t)3 * DM * DM);    // 1536x512
  unsigned short* woutb = carve((size_t)DM * DM);        // 512x512
  unsigned short* regb  = carve((size_t)RR * DM);        // 16x512
  unsigned short* qbuf  = carve((size_t)NH * PP * EE);   // q[h][pix][e]
  unsigned short* kbuf  = carve((size_t)NH * PPP * EE);  // k[h][pix pad][e]
  unsigned short* vTb   = carve((size_t)NH * EE * PPP);  // v[h][e][pix pad]
  unsigned short* regk  = carve((size_t)NH * RR * EE);
  unsigned short* regvT = carve((size_t)NH * EE * RR);
  unsigned short* attn  = carve((size_t)PP * DM);

  // 0) f32 -> bf16
  cvt_f32_bf16<<<(in_sizes[0] + 255) / 256, 256, 0, stream>>>(x, xb, in_sizes[0]);
  cvt_f32_bf16<<<(in_sizes[2] + 255) / 256, 256, 0, stream>>>(wqkv, wqkvb, in_sizes[2]);
  cvt_f32_bf16<<<(in_sizes[3] + 255) / 256, 256, 0, stream>>>(wout, woutb, in_sizes[3]);
  cvt_f32_bf16<<<(in_sizes[1] + 255) / 256, 256, 0, stream>>>(regs, regb, in_sizes[1]);

  // 1) QKV projection (WMMA, 64x64 tiles)
  qkv_gemm<<<dim3(PP / 64, 24), 32, 0, stream>>>(xb, wqkvb, qbuf, kbuf, vTb);

  // 2) register K/V projection (WMMA)
  regkv_gemm<<<16, 32, 0, stream>>>(regb, wqkvb, regk, regvT);

  // 3) neighborhood attention (WMMA logits + WMMA P*V)
  natten_attn<<<dim3(WW / 16, HH, NH), 32, 0, stream>>>(qbuf, kbuf, vTb, regk,
                                                        regvT, attn);

  // 4) output projection (WMMA, 64x64 tiles, f32 out)
  out_gemm<<<dim3(PP / 64, DM / 64), 32, 0, stream>>>(attn, woutb, out);
}
